// Meta_upsample_81930796138533
// MI455X (gfx1250) — compile-verified, matrix operand order checked by reading, not run on hardware
//
#include <hip/hip_runtime.h>
#include <math.h>

typedef __attribute__((ext_vector_type(16))) _Float16 v16h;
typedef __attribute__((ext_vector_type(8)))  float    v8f;

#define BATCH 4
#define CIN   128
#define HIN   96
#define WIN   96
#define H2D   192
#define W2D   192
#define PIX   (H2D * W2D)          // 36864
#define TILES_PER_IMG (PIX / 16)   // 2304
#define NW    4                    // waves (16-pixel tiles) per workgroup

// workspace layout (bytes)
#define WS_COORDS 0
#define WS_ROUT   (PIX * 8)                    // after float2[PIX]
#define WS_WC16   (WS_ROUT + PIX * 16)         // after float4[PIX]
#define WS_WE16   (WS_WC16 + 8192 * 2)         // 8192 halves each

// ---------------------------------------------------------------------------
// Kernel 0: convert weights to f16 in the exact LDS layouts used later
// ---------------------------------------------------------------------------
__global__ __launch_bounds__(256) void convert_weights_kernel(
    const float* __restrict__ wc,   // [E,16,128]
    const float* __restrict__ we,   // [E,128,16]
    _Float16* __restrict__ wc16,    // flat [E][16][128]
    _Float16* __restrict__ we16)    // flat [c][k=(e*16+o)]
{
    int i = blockIdx.x * 256 + threadIdx.x;    // grid: 8192/256 = 32 blocks
    wc16[i] = (_Float16)wc[i];
    int e = i >> 11, c = (i >> 4) & 127, o = i & 15;
    we16[c * 64 + e * 16 + o] = (_Float16)we[i];
}

// ---------------------------------------------------------------------------
// Kernel 1: per-pixel MLP -> absolute sample coords + routing weights
// ---------------------------------------------------------------------------
__global__ __launch_bounds__(256) void routing_kernel(
    const float* __restrict__ w1, const float* __restrict__ b1,
    const float* __restrict__ w2, const float* __restrict__ b2,
    const float* __restrict__ rw, const float* __restrict__ rb,
    const float* __restrict__ ow, const float* __restrict__ ob,
    float2* __restrict__ coords, float4* __restrict__ rout)
{
    __shared__ float sW2[64 * 64];
    int tid = threadIdx.x;
    for (int i = tid; i < 64 * 64; i += 256) sW2[i] = w2[i];
    __syncthreads();

    int p = blockIdx.x * 256 + tid;           // grid sized exactly: PIX/256
    int y = p / W2D, x = p % W2D;
    const float sc = 2.0f;

    float chv = (y + 0.5f) / sc; chv = chv - floorf(chv + 0.001f) - 0.5f;
    float cwv = (x + 0.5f) / sc; cwv = cwv - floorf(cwv + 0.001f) - 0.5f;
    float in0 = 1.0f / sc, in1 = chv, in2 = cwv;

    float h1[64];
#pragma unroll
    for (int j = 0; j < 64; ++j) {
        float s = fmaf(w1[j*3+0], in0, fmaf(w1[j*3+1], in1, fmaf(w1[j*3+2], in2, b1[j])));
        h1[j] = fmaxf(s, 0.0f);
    }

    // Fuse layer-2 with the 6 output dot-products: no h2 array -> no scratch.
    float a_r0 = rb[0], a_r1 = rb[1], a_r2 = rb[2], a_r3 = rb[3];
    float a_o0 = ob[0], a_o1 = ob[1];
    for (int j = 0; j < 64; ++j) {
        float s = b2[j];
#pragma unroll
        for (int k = 0; k < 64; ++k) s = fmaf(sW2[j*64 + k], h1[k], s);
        float h2j = fmaxf(s, 0.0f);
        a_r0 = fmaf(rw[0*64 + j], h2j, a_r0);
        a_r1 = fmaf(rw[1*64 + j], h2j, a_r1);
        a_r2 = fmaf(rw[2*64 + j], h2j, a_r2);
        a_r3 = fmaf(rw[3*64 + j], h2j, a_r3);
        a_o0 = fmaf(ow[0*64 + j], h2j, a_o0);
        a_o1 = fmaf(ow[1*64 + j], h2j, a_o1);
    }

    float4 r;
    r.x = 1.0f / (1.0f + expf(-a_r0));
    r.y = 1.0f / (1.0f + expf(-a_r1));
    r.z = 1.0f / (1.0f + expf(-a_r2));
    r.w = 1.0f / (1.0f + expf(-a_r3));

    // absolute (un-normalized) sample coords: base + learned offset
    float2 c2;
    c2.x = ((x + 0.5f) / sc - 0.5f) + a_o0;
    c2.y = ((y + 0.5f) / sc - 0.5f) + a_o1;
    coords[p] = c2;
    rout[p]   = r;
}

// v16h fragment packing (ISA 7.12.2, 16-bit A 16x32 / B 32x16):
// lanes 0-15 hold K = {0..7, 16..23}, lanes 16-31 hold K = {8..15, 24..31}
__device__ __forceinline__ int kmap(int j, int half) {
    return ((j < 8) ? 0 : 16) + half * 8 + (j & 7);
}

// ---------------------------------------------------------------------------
// Kernel 2: fused bilinear gather + routed compress/expand (WMMA) + residual
// One wave = one 16-pixel row tile of one batch image.
// ---------------------------------------------------------------------------
__global__ __launch_bounds__(32 * NW) void fused_kernel(
    const float*    __restrict__ x,
    const _Float16* __restrict__ wc16,   // flat [E][16][128]
    const _Float16* __restrict__ we16,   // flat [c][64]
    const float2*   __restrict__ coords,
    const float4*   __restrict__ rout,
    float* __restrict__ out)
{
    __shared__ _Float16 WcA[4][16][128];       // 16 KB  A-side compress weights
    __shared__ _Float16 WeA[128][64];          // 16 KB  A-side expand weights (k=(e,o))
    __shared__ float    fea[NW][128][16];      // 32 KB  gathered fea0 tile (f32, residual)
    __shared__ _Float16 compB[NW][64][16];     //  8 KB  routed comp, B-ready

    const int tid = threadIdx.x;

    // cooperative weight staging: straight 16B copies (pre-converted layouts)
    {
        const uint4* s0 = (const uint4*)wc16;
        const uint4* s1 = (const uint4*)we16;
        uint4* d0 = (uint4*)&WcA[0][0][0];
        uint4* d1 = (uint4*)&WeA[0][0];
        for (int i = tid; i < 1024; i += 32 * NW) d0[i] = s0[i];
        for (int i = tid; i < 1024; i += 32 * NW) d1[i] = s1[i];
    }
    __syncthreads();

    const int wave = tid >> 5, lane = tid & 31;
    const int px = lane & 15, half = lane >> 4;

    const int task = blockIdx.x * NW + wave;
    const int b    = task / TILES_PER_IMG;
    const int tile = task % TILES_PER_IMG;
    const int pb   = tile * 16;                 // 16 consecutive pixels in a row
    const int y2   = pb / W2D, x20 = pb % W2D;
    const int pglob = pb + px;

    const float2 crd = coords[pglob];
    const float4 rr  = rout[pglob];
    const float r4[4] = { rr.x, rr.y, rr.z, rr.w };

    // ---- bilinear gather (zeros padding, align_corners) into LDS ----
    {
        float ix = crd.x, iy = crd.y;
        float ix0f = floorf(ix), iy0f = floorf(iy);
        float tx = ix - ix0f, ty = iy - iy0f;
        int ix0 = (int)ix0f, iy0 = (int)iy0f;

        int   offs[4];
        float wm[4];
        const float tw[4] = { (1.f-tx)*(1.f-ty), tx*(1.f-ty), (1.f-tx)*ty, tx*ty };
#pragma unroll
        for (int t = 0; t < 4; ++t) {
            int xi = ix0 + (t & 1);
            int yi = iy0 + (t >> 1);
            bool valid = (xi >= 0) && (xi < WIN) && (yi >= 0) && (yi < HIN);
            int xc = min(max(xi, 0), WIN - 1);
            int yc = min(max(yi, 0), HIN - 1);
            offs[t] = yc * WIN + xc;
            wm[t]   = valid ? tw[t] : 0.0f;
        }
        const float* xb = x + (size_t)b * CIN * HIN * WIN;
        // warm the cache lines for this lane's taps (global_prefetch_b8)
        __builtin_prefetch(xb + (size_t)half * (HIN * WIN) + offs[0], 0, 3);
        __builtin_prefetch(xb + (size_t)half * (HIN * WIN) + offs[2], 0, 3);
        for (int t = 0; t < 64; ++t) {
            int c = 2 * t + half;                       // lanes 0-15: even ch, 16-31: odd ch
            const float* ch = xb + (size_t)c * (HIN * WIN);
            float v = ch[offs[0]] * wm[0] + ch[offs[1]] * wm[1]
                    + ch[offs[2]] * wm[2] + ch[offs[3]] * wm[3];
            fea[wave][c][px] = v;
        }
    }
    asm volatile("" ::: "memory");   // keep LDS write->read order (same wave, DS in-order)

    // ---- B fragments for compress: fea[c][p], K = channel (4 chunks of 32) ----
    v16h Bf[4];
#pragma unroll
    for (int kc = 0; kc < 4; ++kc) {
#pragma unroll
        for (int j = 0; j < 16; ++j)
            Bf[kc][j] = (_Float16)fea[wave][kc * 32 + kmap(j, half)][px];
    }

    // ---- compress: comp[o,p] = sum_e sum_c Wc[e,o,c] * (r[p,e]*fea[c,p]) ----
    v8f comp = {};
#pragma unroll
    for (int e = 0; e < 4; ++e) {
        const _Float16 reh = (_Float16)r4[e];
        v16h rsplat;
#pragma unroll
        for (int j = 0; j < 16; ++j) rsplat[j] = reh;
#pragma unroll
        for (int kc = 0; kc < 4; ++kc) {
            v16h Af;
#pragma unroll
            for (int j = 0; j < 16; ++j)
                Af[j] = WcA[e][px][kc * 32 + kmap(j, half)];  // A: M = o = lane&15
            v16h Bs = Bf[kc] * rsplat;                        // v_pk_mul_f16, r lane-uniform
            comp = __builtin_amdgcn_wmma_f32_16x16x32_f16(
                false, Af, false, Bs, (short)0, comp, false, false);
        }
    }

    // ---- stage routed comp to LDS in B layout: compB[k=(e,o)][p] = r[p,e]*comp[o,p]
    // D layout: lane -> p = lane&15, rows o = i + 8*half
#pragma unroll
    for (int e = 0; e < 4; ++e) {
#pragma unroll
        for (int i = 0; i < 8; ++i)
            compB[wave][e * 16 + i + 8 * half][px] = (_Float16)(comp[i] * r4[e]);
    }
    asm volatile("" ::: "memory");

    v16h Bx[2];
#pragma unroll
    for (int kx = 0; kx < 2; ++kx) {
#pragma unroll
        for (int j = 0; j < 16; ++j)
            Bx[kx][j] = compB[wave][kx * 32 + kmap(j, half)][px];
    }

    // ---- expand: out[c,p] = sum_k We[c,k] * compB[k,p], 8 channel tiles ----
    const size_t obase = ((size_t)b * CIN) * (size_t)(H2D * W2D)
                       + (size_t)y2 * W2D + x20;
#pragma unroll
    for (int mt = 0; mt < 8; ++mt) {
        v8f acc = {};
#pragma unroll
        for (int kx = 0; kx < 2; ++kx) {
            v16h Af;
#pragma unroll
            for (int j = 0; j < 16; ++j)
                Af[j] = WeA[mt * 16 + px][kx * 32 + kmap(j, half)];
            acc = __builtin_amdgcn_wmma_f32_16x16x32_f16(
                false, Af, false, Bx[kx], (short)0, acc, false, false);
        }
        // D: lane -> p, channel = mt*16 + i + 8*half; residual add + store
#pragma unroll
        for (int i = 0; i < 8; ++i) {
            int cch = mt * 16 + i + 8 * half;
            out[obase + (size_t)cch * (H2D * W2D) + px] = acc[i] + fea[wave][cch][px];
        }
    }
}

// ---------------------------------------------------------------------------
extern "C" void kernel_launch(void* const* d_in, const int* in_sizes, int n_in,
                              void* d_out, int out_size, void* d_ws, size_t ws_size,
                              hipStream_t stream) {
    const float* x  = (const float*)d_in[0];
    const float* wc = (const float*)d_in[1];
    const float* we = (const float*)d_in[2];
    const float* w1 = (const float*)d_in[3];
    const float* b1 = (const float*)d_in[4];
    const float* w2 = (const float*)d_in[5];
    const float* b2 = (const float*)d_in[6];
    const float* rw = (const float*)d_in[7];
    const float* rb = (const float*)d_in[8];
    const float* ow = (const float*)d_in[9];
    const float* ob = (const float*)d_in[10];
    // scale is fixed at 2 by the reference setup; geometry is compiled in.

    char* ws = (char*)d_ws;
    float2*   coords = (float2*)(ws + WS_COORDS);
    float4*   routw  = (float4*)(ws + WS_ROUT);
    _Float16* wc16   = (_Float16*)(ws + WS_WC16);
    _Float16* we16   = (_Float16*)(ws + WS_WE16);

    convert_weights_kernel<<<32, 256, 0, stream>>>(wc, we, wc16, we16);
    routing_kernel<<<PIX / 256, 256, 0, stream>>>(w1, b1, w2, b2, rw, rb, ow, ob,
                                                  coords, routw);
    fused_kernel<<<(BATCH * TILES_PER_IMG) / NW, 32 * NW, 0, stream>>>(
        x, wc16, we16, coords, routw, (float*)d_out);
}
